// MambaLayer_19241453486086
// MI455X (gfx1250) — compile-verified, matrix operand order checked by reading, not run on hardware
//
#include <hip/hip_runtime.h>
#include <math.h>

typedef __attribute__((ext_vector_type(16))) __bf16 v16bf;
typedef __attribute__((ext_vector_type(8)))  __bf16 v8bf;
typedef __attribute__((ext_vector_type(8)))  float  v8f;

#define L_SEQ 2048
#define C_DIM 384
#define DIN_  768
#define NST   16
#define RRANK 24
#define XZLD  1536   // 2*DIN
#define PROJLD 64    // padded R+2N (56 -> 64)

// ---------- helpers ----------
__device__ __forceinline__ __bf16 f2bf(float f) {
  union { float f; unsigned u; } a; a.f = f;
  unsigned r = a.u + 0x7FFFu + ((a.u >> 16) & 1u);
  union { unsigned short s; __bf16 b; } o; o.s = (unsigned short)(r >> 16);
  return o.b;
}

__device__ __forceinline__ int perm_s(int l, int perm) {
  // map token index l (in permuted order) to x's flat spatial index s
  if (perm == 0) return l;                                   // (d,w,h)
  if (perm == 1) {                                           // (w,d,h)
    int w = l >> 7, d = (l >> 3) & 15, h = l & 7;
    return (d << 7) | (w << 3) | h;
  }
  int h = l >> 8, w = (l >> 4) & 15, d = l & 15;             // (h,w,d)
  return (d << 7) | (w << 3) | h;
}

// ---------- out = x (init) ----------
__global__ void copy_kernel(const float* __restrict__ src, float* __restrict__ dst, int n) {
  int i = blockIdx.x * blockDim.x + threadIdx.x;
  if (i < n) dst[i] = src[i];
}

// ---------- weight convert + pad to bf16 ----------
__global__ void convert_pad_kernel(const float* __restrict__ src, __bf16* __restrict__ dst,
                                   int N, int K, int Np, int Kp) {
  int i = blockIdx.x * blockDim.x + threadIdx.x;
  if (i >= Np * Kp) return;
  int n = i / Kp, k = i - n * Kp;
  dst[i] = (n < N && k < K) ? f2bf(src[n * K + k]) : f2bf(0.0f);
}

// ---------- permute + layernorm -> bf16 u (L x C) : one wave per token ----------
__global__ void perm_ln_kernel(const float* __restrict__ x, const float* __restrict__ g,
                               const float* __restrict__ b, __bf16* __restrict__ u, int perm) {
  int wave = (blockIdx.x * blockDim.x + threadIdx.x) >> 5;
  int lane = threadIdx.x & 31;
  if (wave >= L_SEQ) return;
  int s = perm_s(wave, perm);
  float vals[12];
  float sum = 0.f;
#pragma unroll
  for (int i = 0; i < 12; ++i) {
    int c = lane * 12 + i;                 // 32*12 = 384
    float v = x[c * L_SEQ + s];
    vals[i] = v; sum += v;
  }
#pragma unroll
  for (int o = 16; o >= 1; o >>= 1) sum += __shfl_xor(sum, o);
  float mean = sum * (1.0f / 384.0f);
  float var = 0.f;
#pragma unroll
  for (int i = 0; i < 12; ++i) { float d = vals[i] - mean; var += d * d; }
#pragma unroll
  for (int o = 16; o >= 1; o >>= 1) var += __shfl_xor(var, o);
  float rstd = rsqrtf(var * (1.0f / 384.0f) + 1e-5f);
#pragma unroll
  for (int i = 0; i < 12; ++i) {
    int c = lane * 12 + i;
    u[wave * C_DIM + c] = f2bf((vals[i] - mean) * rstd * g[c] + b[c]);
  }
}

// ---------- bf16 WMMA GEMM: C[M,N] = A[M,K] * B[N,K]^T ----------
// One wave computes a 16x64 tile: one A fragment reused across 4 B fragments
// (4 accumulators), halving load traffic per v_wmma vs a 16x16 tile.
// A row-major lda=K, B row-major (weight rows = x @ w.T form), C row-major ldc=N.
// Requires: K % 32 == 0, N % 64 == 0, 16B-aligned pointers.
__global__ void gemm_bf16_kernel(const __bf16* __restrict__ A, const __bf16* __restrict__ B,
                                 float* __restrict__ C, int M, int N, int K) {
  int wave = (blockIdx.x * blockDim.x + threadIdx.x) >> 5;
  int lane = threadIdx.x & 31;
  int ntiles = N >> 6;                     // 64-wide output tiles
  int tm = wave / ntiles;
  int tn = wave - tm * ntiles;
  if (tm >= (M >> 4)) return;
  int row  = lane & 15;
  int koff = (lane >> 4) << 3;             // 0 or 8 (ISA 16-bit A-matrix layout)
  const __bf16* Ap = A + (size_t)(tm * 16 + row) * K + koff;
  const __bf16* Bp = B + (size_t)(tn * 64 + row) * K + koff;
  v8f acc0 = {}, acc1 = {}, acc2 = {}, acc3 = {};
  for (int k0 = 0; k0 < K; k0 += 32) {
    union { v16bf v; v8bf h[2]; } af, b0, b1, b2, b3;
    af.h[0] = *(const v8bf*)(Ap + k0);
    af.h[1] = *(const v8bf*)(Ap + k0 + 16);
    b0.h[0] = *(const v8bf*)(Bp + k0);
    b0.h[1] = *(const v8bf*)(Bp + k0 + 16);
    b1.h[0] = *(const v8bf*)(Bp + 16 * K + k0);
    b1.h[1] = *(const v8bf*)(Bp + 16 * K + k0 + 16);
    b2.h[0] = *(const v8bf*)(Bp + 32 * K + k0);
    b2.h[1] = *(const v8bf*)(Bp + 32 * K + k0 + 16);
    b3.h[0] = *(const v8bf*)(Bp + 48 * K + k0);
    b3.h[1] = *(const v8bf*)(Bp + 48 * K + k0 + 16);
    // hint next k-block toward the caches (global_prefetch_b8)
    __builtin_prefetch(Ap + k0 + 32, 0, 0);
    __builtin_prefetch(Bp + k0 + 32, 0, 0);
    acc0 = __builtin_amdgcn_wmma_f32_16x16x32_bf16(false, af.v, false, b0.v, (short)0, acc0, false, false);
    acc1 = __builtin_amdgcn_wmma_f32_16x16x32_bf16(false, af.v, false, b1.v, (short)0, acc1, false, false);
    acc2 = __builtin_amdgcn_wmma_f32_16x16x32_bf16(false, af.v, false, b2.v, (short)0, acc2, false, false);
    acc3 = __builtin_amdgcn_wmma_f32_16x16x32_bf16(false, af.v, false, b3.v, (short)0, acc3, false, false);
  }
  // D layout: lane 0-15 -> N=lane, M=vgpr; lane 16-31 -> N=lane-16, M=8+vgpr
  int crow = (lane >> 4) * 8;
  float* Cp = C + (size_t)(tm * 16 + crow) * N + tn * 64 + (lane & 15);
#pragma unroll
  for (int r = 0; r < 8; ++r) {
    Cp[(size_t)r * N]      = acc0[r];
    Cp[(size_t)r * N + 16] = acc1[r];
    Cp[(size_t)r * N + 32] = acc2[r];
    Cp[(size_t)r * N + 48] = acc3[r];
  }
}

// ---------- depthwise causal conv (K=4) + SiLU; rev=1 -> anti-causal taps ----------
__global__ void conv_silu_kernel(const float* __restrict__ xz, const float* __restrict__ w,
                                 const float* __restrict__ b, float* __restrict__ xc,
                                 __bf16* __restrict__ xcb, int rev) {
  int i = blockIdx.x * blockDim.x + threadIdx.x;
  if (i >= L_SEQ * DIN_) return;
  int l = i / DIN_, d = i - l * DIN_;
  float acc = b[d];
#pragma unroll
  for (int k = 0; k < 4; ++k) {
    int t = rev ? (l + 3 - k) : (l - 3 + k);
    float xv = (t >= 0 && t < L_SEQ) ? xz[(size_t)t * XZLD + d] : 0.0f;
    acc += w[d * 4 + k] * xv;
  }
  float sv = acc / (1.0f + __expf(-acc));   // SiLU
  xc[i] = sv;
  xcb[i] = f2bf(sv);
}

// ---------- dt slice (proj[:, :24]) -> bf16, K padded to 32 ----------
__global__ void dtpad_kernel(const float* __restrict__ proj, __bf16* __restrict__ dt) {
  int i = blockIdx.x * blockDim.x + threadIdx.x;
  if (i >= L_SEQ * 32) return;
  int l = i >> 5, j = i & 31;
  dt[i] = (j < RRANK) ? f2bf(proj[l * PROJLD + j]) : f2bf(0.0f);
}

// ---------- selective scan: 16 lanes per channel (states), shfl reduce over N ----------
__global__ void scan_kernel(const float* __restrict__ draw, const float* __restrict__ dtb,
                            const float* __restrict__ xc, const float* __restrict__ proj,
                            const float* __restrict__ A_log, const float* __restrict__ Dp,
                            float* __restrict__ yacc, int rev, int accumulate) {
  int idx = blockIdx.x * blockDim.x + threadIdx.x;
  int d = idx >> 4, n = idx & 15;
  if (d >= DIN_) return;
  float A  = -__expf(A_log[d * NST + n]);
  float db = dtb[d];
  float Dv = Dp[d];
  float h = 0.0f;
  for (int step = 0; step < L_SEQ; ++step) {
    int t = rev ? (L_SEQ - 1 - step) : step;
    float x0 = draw[(size_t)t * DIN_ + d] + db;
    float delta = (x0 > 20.0f) ? x0 : log1pf(__expf(x0));   // softplus
    float xv = xc[(size_t)t * DIN_ + d];
    float bm = proj[t * PROJLD + RRANK + n];        // Bm
    float cm = proj[t * PROJLD + RRANK + NST + n];  // Cm
    h = __expf(delta * A) * h + delta * xv * bm;
    float part = h * cm;
#pragma unroll
    for (int o = 8; o >= 1; o >>= 1) part += __shfl_xor(part, o);
    if (n == 0) {
      float y = part + xv * Dv;
      if (accumulate) yacc[(size_t)t * DIN_ + d] += y;
      else            yacc[(size_t)t * DIN_ + d]  = y;
    }
  }
}

// ---------- (y_f + y_b) * silu(z) -> bf16 ----------
__global__ void gate_kernel(const float* __restrict__ yacc, const float* __restrict__ xz,
                            __bf16* __restrict__ yb) {
  int i = blockIdx.x * blockDim.x + threadIdx.x;
  if (i >= L_SEQ * DIN_) return;
  int l = i / DIN_, d = i - l * DIN_;
  float z = xz[(size_t)l * XZLD + DIN_ + d];
  float s = z / (1.0f + __expf(-z));
  yb[i] = f2bf(yacc[i] * s);
}

// ---------- scatter-add permuted output into d_out ----------
__global__ void scatter_add_kernel(const float* __restrict__ o, float* __restrict__ out, int perm) {
  int i = blockIdx.x * blockDim.x + threadIdx.x;
  if (i >= L_SEQ * C_DIM) return;
  int l = i / C_DIM, c = i - l * C_DIM;
  out[c * L_SEQ + perm_s(l, perm)] += o[i];
}

// ==================== launch ====================
static inline void* ws_alloc(char** cur, size_t bytes) {
  void* p = *cur;
  *cur += (bytes + 255) & ~(size_t)255;
  return p;
}

extern "C" void kernel_launch(void* const* d_in, const int* in_sizes, int n_in,
                              void* d_out, int out_size, void* d_ws, size_t ws_size,
                              hipStream_t stream) {
  (void)in_sizes; (void)n_in; (void)out_size; (void)ws_size;
  const float* x       = (const float*)d_in[0];
  const float* ln_g    = (const float*)d_in[1];
  const float* ln_b    = (const float*)d_in[2];
  const float* in_w    = (const float*)d_in[3];
  const float* out_w   = (const float*)d_in[4];
  const float* conv_w[2] = {(const float*)d_in[5],  (const float*)d_in[12]};
  const float* conv_b[2] = {(const float*)d_in[6],  (const float*)d_in[13]};
  const float* xp_w[2]   = {(const float*)d_in[7],  (const float*)d_in[14]};
  const float* dt_w[2]   = {(const float*)d_in[8],  (const float*)d_in[15]};
  const float* dt_b[2]   = {(const float*)d_in[9],  (const float*)d_in[16]};
  const float* A_log[2]  = {(const float*)d_in[10], (const float*)d_in[17]};
  const float* Dp[2]     = {(const float*)d_in[11], (const float*)d_in[18]};
  float* out = (float*)d_out;

  char* cur = (char*)d_ws;
  __bf16* u_bf    = (__bf16*)ws_alloc(&cur, (size_t)L_SEQ * C_DIM * 2);
  float*  xz      = (float*) ws_alloc(&cur, (size_t)L_SEQ * XZLD * 4);
  float*  xc      = (float*) ws_alloc(&cur, (size_t)L_SEQ * DIN_ * 4);
  __bf16* xc_bf   = (__bf16*)ws_alloc(&cur, (size_t)L_SEQ * DIN_ * 2);
  float*  proj    = (float*) ws_alloc(&cur, (size_t)L_SEQ * PROJLD * 4);
  __bf16* dt_bf   = (__bf16*)ws_alloc(&cur, (size_t)L_SEQ * 32 * 2);
  float*  draw    = (float*) ws_alloc(&cur, (size_t)L_SEQ * DIN_ * 4);
  float*  yacc    = (float*) ws_alloc(&cur, (size_t)L_SEQ * DIN_ * 4);
  __bf16* y_bf    = (__bf16*)ws_alloc(&cur, (size_t)L_SEQ * DIN_ * 2);
  float*  o_p     = (float*) ws_alloc(&cur, (size_t)L_SEQ * C_DIM * 4);
  __bf16* w_in_bf  = (__bf16*)ws_alloc(&cur, (size_t)XZLD * C_DIM * 2);
  __bf16* w_out_bf = (__bf16*)ws_alloc(&cur, (size_t)C_DIM * DIN_ * 2);
  __bf16* w_xp_bf[2], *w_dt_bf[2];
  for (int d = 0; d < 2; ++d) {
    w_xp_bf[d] = (__bf16*)ws_alloc(&cur, (size_t)PROJLD * DIN_ * 2);
    w_dt_bf[d] = (__bf16*)ws_alloc(&cur, (size_t)DIN_ * 32 * 2);
  }

  const int T = 256;
  auto blks = [](long n, int t) { return (int)((n + t - 1) / t); };

  // d_out = x
  copy_kernel<<<blks((long)C_DIM * L_SEQ, T), T, 0, stream>>>(x, out, C_DIM * L_SEQ);

  // weights -> bf16 (padded)
  convert_pad_kernel<<<blks((long)XZLD * C_DIM, T), T, 0, stream>>>(in_w,  w_in_bf,  XZLD, C_DIM, XZLD, C_DIM);
  convert_pad_kernel<<<blks((long)C_DIM * DIN_, T), T, 0, stream>>>(out_w, w_out_bf, C_DIM, DIN_, C_DIM, DIN_);
  for (int d = 0; d < 2; ++d) {
    convert_pad_kernel<<<blks((long)PROJLD * DIN_, T), T, 0, stream>>>(xp_w[d], w_xp_bf[d], RRANK + 2 * NST, DIN_, PROJLD, DIN_);
    convert_pad_kernel<<<blks((long)DIN_ * 32, T), T, 0, stream>>>(dt_w[d], w_dt_bf[d], DIN_, RRANK, DIN_, 32);
  }

  auto gemm = [&](const __bf16* A, const __bf16* B, float* Cm, int M, int N, int K) {
    long waves = (long)(M >> 4) * (N >> 6);   // 16x64 tile per wave
    gemm_bf16_kernel<<<blks(waves * 32, T), T, 0, stream>>>(A, B, Cm, M, N, K);
  };

  for (int p = 0; p < 3; ++p) {
    perm_ln_kernel<<<blks((long)L_SEQ * 32, T), T, 0, stream>>>(x, ln_g, ln_b, u_bf, p);
    gemm(u_bf, w_in_bf, xz, L_SEQ, XZLD, C_DIM);                       // in_proj
    for (int dir = 0; dir < 2; ++dir) {
      conv_silu_kernel<<<blks((long)L_SEQ * DIN_, T), T, 0, stream>>>(xz, conv_w[dir], conv_b[dir], xc, xc_bf, dir);
      gemm(xc_bf, w_xp_bf[dir], proj, L_SEQ, PROJLD, DIN_);            // x_proj
      dtpad_kernel<<<blks((long)L_SEQ * 32, T), T, 0, stream>>>(proj, dt_bf);
      gemm(dt_bf, w_dt_bf[dir], draw, L_SEQ, DIN_, 32);                // dt_proj (raw)
      scan_kernel<<<blks((long)DIN_ * NST, T), T, 0, stream>>>(draw, dt_b[dir], xc, proj,
                                                               A_log[dir], Dp[dir], yacc, dir, dir);
    }
    gate_kernel<<<blks((long)L_SEQ * DIN_, T), T, 0, stream>>>(yacc, xz, y_bf);
    gemm(y_bf, w_out_bf, o_p, L_SEQ, C_DIM, DIN_);                     // out_proj
    scatter_add_kernel<<<blks((long)L_SEQ * C_DIM, T), T, 0, stream>>>(o_p, out, p);
  }
}
